// MultiHeadAttention_81363860455568
// MI455X (gfx1250) — compile-verified, matrix operand order checked by reading, not run on hardware
//
#include <hip/hip_runtime.h>
#include <hip/hip_bf16.h>

// ---------------------------------------------------------------------------
// MI455X (gfx1250) attention pipeline, all matmuls via v_wmma_f32_16x16x32_bf16
// Strategy: one elementwise fp32->bf16 downconvert pass over activations and
// weights, then every GEMM is pure bf16 WMMA (no conversion in hot loops).
// ---------------------------------------------------------------------------

typedef __attribute__((ext_vector_type(16))) __bf16    v16bf;
typedef __attribute__((ext_vector_type(8)))  float     v8f;
typedef __attribute__((ext_vector_type(8)))  unsigned  u32x8;

// Pointer types for the gfx1250 async global->LDS copy builtin
// (param type per hipcc diagnostic: vector of 4 ints, global AS, non-const).
typedef int v4i_ __attribute__((vector_size(16)));
typedef __attribute__((address_space(1))) v4i_* gas_v4i_ptr;
typedef __attribute__((address_space(3))) v4i_* las_v4i_ptr;

#define B_  2u
#define S_  2048u
#define D_  1024u
#define H_  16u
#define DK_ 64u

// Pack two floats into one u32 holding two bf16 (RNE bit trick; gfx1250 has
// no v_cvt_pk_bf16_f32 -- probed via __has_builtin in round 2).
__device__ __forceinline__ unsigned pk2bf(float lo, float hi) {
    unsigned ulo = __float_as_uint(lo);
    unsigned uhi = __float_as_uint(hi);
    unsigned rlo = (ulo + 0x7FFFu + ((ulo >> 16) & 1u)) >> 16;
    unsigned rhi = (uhi + 0x7FFFu + ((uhi >> 16) & 1u)) >> 16;
    return rlo | (rhi << 16);
}
__device__ __forceinline__ unsigned short f2bf_bits(float f) {
    unsigned u = __float_as_uint(f);
    return (unsigned short)((u + 0x7FFFu + ((u >> 16) & 1u)) >> 16);
}
__device__ __forceinline__ float bf2f(unsigned short h) {
    return __uint_as_float(((unsigned)h) << 16);
}

// One 16-element WMMA fragment slice from bf16 (raw u16) memory.
// ISA 16-bit A-layout (mirrored for B with lane = N): per lane the 16 K
// values are { half*8 + 0..7 , half*8 + 16..23 } relative to the k-base.
// Two b128 loads, zero data movement.
__device__ __forceinline__ v16bf load_frag_bf16(const unsigned short* __restrict__ p, unsigned half) {
    uint4 lo = *(const uint4*)(p + half * 8);
    uint4 hi = *(const uint4*)(p + half * 8 + 16);
    u32x8 t = {lo.x, lo.y, lo.z, lo.w, hi.x, hi.y, hi.z, hi.w};
    return __builtin_bit_cast(v16bf, t);
}

// ---------------------------------------------------------------------------
// Elementwise fp32 -> bf16 downconvert, 8 elements/thread (b128 in, b128 out).
// ---------------------------------------------------------------------------
__global__ void __launch_bounds__(256) cvt_kernel(const float* __restrict__ X,
                                                  unsigned short* __restrict__ Y,
                                                  unsigned n8) {
    unsigned i = blockIdx.x * 256u + threadIdx.x;
    if (i >= n8) return;
    const float4* p = (const float4*)(X + (size_t)i * 8);
    float4 a = p[0], b = p[1];
    uint4 o = {pk2bf(a.x, a.y), pk2bf(a.z, a.w), pk2bf(b.x, b.y), pk2bf(b.z, b.w)};
    *(uint4*)(Y + (size_t)i * 8) = o;
}

// ---------------------------------------------------------------------------
// contrib[b,s] = (rs - r)/(rs + 1e-9) * (1 - r) * 100,  rs = sum_s reaches[b,s]
// ---------------------------------------------------------------------------
__global__ void __launch_bounds__(256) contrib_kernel(const float* __restrict__ reaches,
                                                      float* __restrict__ contrib) {
    __shared__ float red[256];
    unsigned b = blockIdx.x;
    float s = 0.f;
    for (unsigned i = threadIdx.x; i < S_; i += 256) s += reaches[b * S_ + i];
    red[threadIdx.x] = s;
    __syncthreads();
#pragma unroll
    for (unsigned st = 128; st > 0; st >>= 1) {
        if (threadIdx.x < st) red[threadIdx.x] += red[threadIdx.x + st];
        __syncthreads();
    }
    float rs = red[0];
    for (unsigned i = threadIdx.x; i < S_; i += 256) {
        float r = reaches[b * S_ + i];
        contrib[b * S_ + i] = (rs - r) / (rs + 1e-9f) * (1.0f - r) * 100.0f;
    }
}

// ---------------------------------------------------------------------------
// GEMM: C[m,n] = sum_k A[m,k] * Bw[n,k]   (== A @ Bw.T), both operands bf16.
// VMODE 0: C bf16 row-major [M,N]
// VMODE 1: C bf16 V-transposed per-head layout Vt[b,h,dk,S]
// VMODE 2: C fp32 row-major [M,N]
// Block = 8 waves, each wave owns one 16x16 tile (block tile 128 x 16).
// ---------------------------------------------------------------------------
template <int VMODE>
__global__ void __launch_bounds__(256) gemm_bf16_kernel(const unsigned short* __restrict__ A,
                                                        const unsigned short* __restrict__ Bw,
                                                        void* __restrict__ Out,
                                                        unsigned M, unsigned N, unsigned K) {
    unsigned wave = threadIdx.x >> 5;
    unsigned lane = threadIdx.x & 31;
    unsigned half = lane >> 4;
    unsigned lq   = lane & 15;
    unsigned m0 = blockIdx.x * 128 + wave * 16;
    unsigned n0 = blockIdx.y * 16;

    const unsigned short* arow = A  + (size_t)(m0 + lq) * K;   // A operand: lane = M
    const unsigned short* brow = Bw + (size_t)(n0 + lq) * K;   // B operand: lane = N

    v8f acc = {};
#pragma unroll 4
    for (unsigned kb = 0; kb < K; kb += 32) {
        __builtin_prefetch(arow + kb + 512, 0, 3);   // global_prefetch_b8, WGP scope
        __builtin_prefetch(brow + kb + 512, 0, 3);
        v16bf a = load_frag_bf16(arow + kb, half);
        v16bf b = load_frag_bf16(brow + kb, half);
        acc = __builtin_amdgcn_wmma_f32_16x16x32_bf16(false, a, false, b,
                                                      (short)0, acc, false, false);
    }
#pragma unroll
    for (unsigned r = 0; r < 8; ++r) {
        unsigned m = m0 + r + 8 * half;  // C layout: M = r + 8*(lane>=16), N = lane&15
        unsigned n = n0 + lq;
        if (VMODE == 0) {
            ((unsigned short*)Out)[(size_t)m * N + n] = f2bf_bits(acc[r]);
        } else if (VMODE == 1) {
            unsigned b_ = m / S_, s = m % S_;     // constants: shift/and only
            unsigned h  = n / DK_, dk = n % DK_;
            ((unsigned short*)Out)[(((size_t)(b_ * H_ + h)) * DK_ + dk) * S_ + s] = f2bf_bits(acc[r]);
        } else {
            ((float*)Out)[(size_t)m * N + n] = acc[r];
        }
    }
}

// ---------------------------------------------------------------------------
// Flash-style attention core. One wave = 16 query rows of one (b,h).
// scores = qh @ kh^T / 8; full-row softmax; post-softmax reach & diagonal
// (1e-6) weighting; out = (vh - P@vh) * contrib; bf16 concat output.
// ---------------------------------------------------------------------------
__global__ void __launch_bounds__(128) attn_kernel(const unsigned short* __restrict__ Qb,
                                                   const unsigned short* __restrict__ Kb,
                                                   const unsigned short* __restrict__ Vt,
                                                   const float* __restrict__ reaches,
                                                   const float* __restrict__ contrib,
                                                   unsigned short* __restrict__ Yout) {
    __shared__ float sReach[S_];                 // 8 KB
    __shared__ unsigned short sP[4][16 * 32];    // 4 KB, per-wave P tile

    unsigned wave = threadIdx.x >> 5;
    unsigned lane = threadIdx.x & 31;
    unsigned half = lane >> 4;
    unsigned lq   = lane & 15;

    unsigned bid  = blockIdx.x;          // B * H * (S/64)
    unsigned qblk = bid % (S_ / 64);
    unsigned bh   = bid / (S_ / 64);
    unsigned h    = bh % H_;
    unsigned b    = bh / H_;
    unsigned q0   = qblk * 64 + wave * 16;

    // Stage reaches[b,:] (8 KB) into LDS via the CDNA5 async-DMA path when the
    // toolchain exposes it (ASYNCcnt-tracked, no VGPR round trip).
#if __has_builtin(__builtin_amdgcn_global_load_async_to_lds_b128)
    {
        const float* gsrc = reaches + b * S_;
        for (unsigned i = threadIdx.x; i < S_ / 4; i += 128) {
            __builtin_amdgcn_global_load_async_to_lds_b128(
                (gas_v4i_ptr)(gsrc + i * 4),
                (las_v4i_ptr)(&sReach[i * 4]),
                0, 0);
        }
        asm volatile("s_wait_asynccnt 0x0" ::: "memory");
    }
#else
    for (unsigned i = threadIdx.x; i < S_; i += 128) sReach[i] = reaches[b * S_ + i];
#endif
    __syncthreads();

    // Q fragments (A operand, lane = query row), dk chunks 0..31 / 32..63
    const unsigned short* qrow = Qb + ((size_t)(b * S_ + q0 + lq)) * D_ + h * DK_;
    v16bf aq0 = load_frag_bf16(qrow, half);
    v16bf aq1 = load_frag_bf16(qrow + 32, half);

    v8f acc[4] = {};                 // out tile 16 x 64 (four N-chunks of 16)
    float mrow[8], lrow[8];
#pragma unroll
    for (int r = 0; r < 8; ++r) { mrow[r] = -1e30f; lrow[r] = 0.f; }

    const unsigned short* kbase = Kb + ((size_t)(b * S_)) * D_ + h * DK_;
    const unsigned short* vbase = Vt + (((size_t)(b * H_ + h)) * DK_) * S_;

    for (unsigned kb = 0; kb < S_; kb += 32) {
        // ---- scores: two 16x16 key tiles, K-dim = dk = 64 (2 chunks) ----
        const unsigned short* k0row = kbase + (size_t)(kb + lq) * D_;
        const unsigned short* k1row = kbase + (size_t)(kb + 16 + lq) * D_;
        __builtin_prefetch(k0row + 32 * D_, 0, 3);
        __builtin_prefetch(k1row + 32 * D_, 0, 3);
        v8f s0 = {}, s1 = {};
        s0 = __builtin_amdgcn_wmma_f32_16x16x32_bf16(false, aq0, false,
                 load_frag_bf16(k0row, half), (short)0, s0, false, false);
        s0 = __builtin_amdgcn_wmma_f32_16x16x32_bf16(false, aq1, false,
                 load_frag_bf16(k0row + 32, half), (short)0, s0, false, false);
        s1 = __builtin_amdgcn_wmma_f32_16x16x32_bf16(false, aq0, false,
                 load_frag_bf16(k1row, half), (short)0, s1, false, false);
        s1 = __builtin_amdgcn_wmma_f32_16x16x32_bf16(false, aq1, false,
                 load_frag_bf16(k1row + 32, half), (short)0, s1, false, false);

        // ---- online softmax + post-softmax weighting, P -> LDS (bf16) ----
#pragma unroll
        for (int r = 0; r < 8; ++r) {
            float x0 = s0[r] * 0.125f;        // 1/sqrt(64)
            float x1 = s1[r] * 0.125f;
            float t = fmaxf(x0, x1);
#pragma unroll
            for (int msk = 1; msk <= 8; msk <<= 1) t = fmaxf(t, __shfl_xor(t, msk, 32));
            float mnew  = fmaxf(mrow[r], t);
            float alpha = __expf(mrow[r] - mnew);
            mrow[r] = mnew;
            float p0 = __expf(x0 - mnew);
            float p1 = __expf(x1 - mnew);
            float ps = p0 + p1;
#pragma unroll
            for (int msk = 1; msk <= 8; msk <<= 1) ps += __shfl_xor(ps, msk, 32);
            lrow[r] = lrow[r] * alpha + ps;   // unweighted softmax denominator
#pragma unroll
            for (int c = 0; c < 4; ++c) acc[c][r] *= alpha;

            unsigned row = r + 8 * half;
            unsigned qg  = q0 + row;
            unsigned kg0 = kb + lq, kg1 = kb + 16 + lq;
            float w0 = sReach[kg0] * ((qg == kg0) ? 1e-6f : 1.0f);
            float w1 = sReach[kg1] * ((qg == kg1) ? 1e-6f : 1.0f);
            sP[wave][row * 32 + lq]      = f2bf_bits(p0 * w0);
            sP[wave][row * 32 + 16 + lq] = f2bf_bits(p1 * w1);
        }
        asm volatile("s_wait_dscnt 0x0" ::: "memory");

        // ---- P @ V : A = P (16x32 from LDS), B = Vt rows (keys contiguous) ----
        v16bf ap = load_frag_bf16(&sP[wave][lq * 32], half);
#pragma unroll
        for (unsigned c = 0; c < 4; ++c) {
            const unsigned short* vrow = vbase + (size_t)(c * 16 + lq) * S_ + kb;
            __builtin_prefetch(vrow + 32, 0, 3);
            v16bf bv = load_frag_bf16(vrow, half);
            acc[c] = __builtin_amdgcn_wmma_f32_16x16x32_bf16(false, ap, false, bv,
                                                             (short)0, acc[c], false, false);
        }
    }

    // ---- epilogue: out = (vh - acc/l) * contrib, store bf16 concat [B,S,D] ----
    float cb[8];
#pragma unroll
    for (int r = 0; r < 8; ++r) cb[r] = contrib[b * S_ + q0 + r + 8 * half];
#pragma unroll
    for (unsigned c = 0; c < 4; ++c) {
        unsigned dk = c * 16 + lq;
#pragma unroll
        for (int r = 0; r < 8; ++r) {
            unsigned qg = q0 + r + 8 * half;
            float vh = bf2f(vbase[(size_t)dk * S_ + qg]);
            float o  = (vh - acc[c][r] / lrow[r]) * cb[r];
            Yout[((size_t)(b * S_ + qg)) * D_ + h * DK_ + dk] = f2bf_bits(o);
        }
    }
}

// ---------------------------------------------------------------------------
extern "C" void kernel_launch(void* const* d_in, const int* in_sizes, int n_in,
                              void* d_out, int out_size, void* d_ws, size_t ws_size,
                              hipStream_t stream) {
    const float* q       = (const float*)d_in[0];
    const float* k       = (const float*)d_in[1];
    const float* v       = (const float*)d_in[2];
    const float* reaches = (const float*)d_in[3];
    const float* Wq      = (const float*)d_in[4];
    const float* Wk      = (const float*)d_in[5];
    const float* Wv      = (const float*)d_in[6];
    const float* Wo      = (const float*)d_in[7];
    float* out = (float*)d_out;

    const unsigned M   = B_ * S_;                                 // 4096
    const size_t  actb = (size_t)M * D_ * sizeof(unsigned short); // 8 MB
    const size_t  wgtb = (size_t)D_ * D_ * sizeof(unsigned short);// 2 MB

    char* ws = (char*)d_ws;
    unsigned short* Qb  = (unsigned short*)(ws + 0 * actb);   // proj outputs
    unsigned short* Kb  = (unsigned short*)(ws + 1 * actb);
    unsigned short* Vt  = (unsigned short*)(ws + 2 * actb);
    unsigned short* Yb  = (unsigned short*)(ws + 3 * actb);
    unsigned short* qc  = (unsigned short*)(ws + 4 * actb);   // bf16 inputs
    unsigned short* kc  = (unsigned short*)(ws + 5 * actb);
    unsigned short* vc  = (unsigned short*)(ws + 6 * actb);
    unsigned short* Wqc = (unsigned short*)(ws + 7 * actb);
    unsigned short* Wkc = (unsigned short*)(ws + 7 * actb + 1 * wgtb);
    unsigned short* Wvc = (unsigned short*)(ws + 7 * actb + 2 * wgtb);
    unsigned short* Woc = (unsigned short*)(ws + 7 * actb + 3 * wgtb);
    float*          ctr = (float*)(ws + 7 * actb + 4 * wgtb);

    // Phase 0: downconvert activations + weights to bf16 (once), contrib
    const unsigned act8 = M * D_ / 8, wgt8 = D_ * D_ / 8;
    cvt_kernel<<<(act8 + 255) / 256, 256, 0, stream>>>(q,  qc,  act8);
    cvt_kernel<<<(act8 + 255) / 256, 256, 0, stream>>>(k,  kc,  act8);
    cvt_kernel<<<(act8 + 255) / 256, 256, 0, stream>>>(v,  vc,  act8);
    cvt_kernel<<<(wgt8 + 255) / 256, 256, 0, stream>>>(Wq, Wqc, wgt8);
    cvt_kernel<<<(wgt8 + 255) / 256, 256, 0, stream>>>(Wk, Wkc, wgt8);
    cvt_kernel<<<(wgt8 + 255) / 256, 256, 0, stream>>>(Wv, Wvc, wgt8);
    cvt_kernel<<<(wgt8 + 255) / 256, 256, 0, stream>>>(Wo, Woc, wgt8);
    contrib_kernel<<<B_, 256, 0, stream>>>(reaches, ctr);

    // Phase 1: projections (x @ W.T) -> bf16, pure bf16 WMMA
    dim3 gproj(M / 128, D_ / 16);
    gemm_bf16_kernel<0><<<gproj, 256, 0, stream>>>(qc, Wqc, Qb, M, D_, D_);
    gemm_bf16_kernel<0><<<gproj, 256, 0, stream>>>(kc, Wkc, Kb, M, D_, D_);
    gemm_bf16_kernel<1><<<gproj, 256, 0, stream>>>(vc, Wvc, Vt, M, D_, D_);

    // Phase 2: attention core (flash-style, one wave = 16 query rows)
    attn_kernel<<<B_ * H_ * (S_ / 64), 128, 0, stream>>>(Qb, Kb, Vt, reaches, ctr, Yb);

    // Phase 3: output projection (Y @ Wo.T) -> fp32
    gemm_bf16_kernel<2><<<gproj, 256, 0, stream>>>(Yb, Woc, out, M, D_, D_);
}
